// MMDiTBlock_17008070492195
// MI455X (gfx1250) — matrix operand order repaired and checked by reading, not compiled, not run on hardware
//
#include <hip/hip_runtime.h>
#include <hip/hip_bf16.h>
#include <cstdint>
#include <cstddef>

// ---------------------------------------------------------------------------
// MMDiT block for MI455X (gfx1250): bf16 WMMA + TDM async A-tile (activation) staging.
// ---------------------------------------------------------------------------

typedef __bf16 bf16_t;
typedef bf16_t v16bf __attribute__((ext_vector_type(16)));
typedef bf16_t v2bf  __attribute__((ext_vector_type(2)));
typedef float  v8f   __attribute__((ext_vector_type(8)));
typedef unsigned int u32x4 __attribute__((ext_vector_type(4)));
typedef int i32x8 __attribute__((ext_vector_type(8)));
typedef int i32x4 __attribute__((ext_vector_type(4)));

#define DEVI __device__ __forceinline__

#if __has_builtin(__builtin_amdgcn_tensor_load_to_lds) && \
    __has_builtin(__builtin_amdgcn_s_wait_tensorcnt)
#define HAVE_TDM 1
#else
#define HAVE_TDM 0
#endif

// ---- WMMA wrapper: D = A(16x32 bf16) * B(32x16 bf16) + C(16x16 f32) -------
DEVI v8f wmma_bf16(v16bf a, v16bf b, v8f c) {
  return __builtin_amdgcn_wmma_f32_16x16x32_bf16(
      /*neg_a=*/false, a, /*neg_b=*/false, b,
      /*c_mod=*/(short)0, c, /*reuse_a=*/false, /*reuse_b=*/false);
}

// ---- A fragment (16x32, bf16) from row-major LDS tile ---------------------
// lane l: lo=l&15 -> M row, hi=l>>4 selects K half. element pair (2g,2g+1):
//   K = (g&3)*2 + (g>=4 ? 16 : 0) + hi*8 + {0,1}
DEVI v16bf load_frag_a(const bf16_t* sm, int pitch, int m0, int k0, int lane) {
  int lo = lane & 15, hi = lane >> 4;
  const bf16_t* row = sm + (size_t)(m0 + lo) * pitch + k0 + hi * 8;
  v16bf a;
#pragma unroll
  for (int g = 0; g < 8; ++g) {
    int kb = (g & 3) * 2 + ((g >= 4) ? 16 : 0);
    v2bf p = *(const v2bf*)(row + kb);
    a[2 * g] = p[0];
    a[2 * g + 1] = p[1];
  }
  return a;
}

// ---- A fragment from f32 LDS (attention probabilities), cvt to bf16 -------
DEVI v16bf load_frag_a_f32(const float* sm, int pitch, int m0, int k0, int lane) {
  int lo = lane & 15, hi = lane >> 4;
  const float* row = sm + (size_t)(m0 + lo) * pitch + k0 + hi * 8;
  v16bf a;
#pragma unroll
  for (int g = 0; g < 8; ++g) {
    int kb = (g & 3) * 2 + ((g >= 4) ? 16 : 0);
    a[2 * g] = (bf16_t)row[kb];
    a[2 * g + 1] = (bf16_t)row[kb + 1];
  }
  return a;
}

// ---- B fragment (32x16, bf16) from N-MAJOR LDS tile Bs[n][k] --------------
// lane l: n = n0 + (l&15); element pair (2g,2g+1) <-> K = (g&3)*2 + (g>=4 ? 16 : 0) + (l>>4)*8 + {0,1}.
DEVI v16bf load_frag_b(const bf16_t* sm, int pitch, int n0, int k0, int lane) {
  int lo = lane & 15, hi = lane >> 4;
  const bf16_t* row = sm + (size_t)(n0 + lo) * pitch + k0 + hi * 8;
  v16bf b;
#pragma unroll
  for (int g = 0; g < 8; ++g) {
    v2bf p = *(const v2bf*)(row + (g & 3) * 2 + ((g >= 4) ? 16 : 0));
    b[2 * g] = p[0];
    b[2 * g + 1] = p[1];
  }
  return b;
}

DEVI float wave_max(float v) {
#pragma unroll
  for (int o = 16; o; o >>= 1) v = fmaxf(v, __shfl_xor(v, o, 32));
  return v;
}
DEVI float wave_sum(float v) {
#pragma unroll
  for (int o = 16; o; o >>= 1) v += __shfl_xor(v, o, 32);
  return v;
}

// ---------------------------------------------------------------------------
// TDM: async DMA of a (tileM x tileK) bf16 tile (row stride = K elements)
// from global into LDS at byte offset ldsOff.  LDS padding is configured as
// pad-4-DWORDs every 16 DWORDs => destination pitch 32+8 = 40 bf16 (GEMM_PA).
// ---------------------------------------------------------------------------
DEVI void tdm_load_a_tile(unsigned ldsOff, const bf16_t* g, int K,
                          int tileK, int tileM) {
#if HAVE_TDM
  uint64_t ga = (uint64_t)(uintptr_t)g;
  u32x4 g0;
  g0[0] = 1u;                                               // count=1 (valid)
  g0[1] = ldsOff;                                           // lds_addr
  g0[2] = (unsigned)ga;                                     // global_addr lo
  g0[3] = (unsigned)((ga >> 32) & 0x1FFFFFFu) | (2u << 30); // ga hi | type=2
  unsigned td0 = (unsigned)K;          // tensor_dim0 (elements)
  unsigned td1 = 1u << 20;             // tensor_dim1 (rows, generous)
  uint64_t s0 = (uint64_t)(unsigned)K; // tensor_dim0_stride (elements)
  i32x8 g1;
  g1[0] = (int)((1u << 16)             // data_size = 2 bytes
              | (1u << 20)             // pad_enable
              | (3u << 22)             // pad_interval: every 16 DWORDs
              | (3u << 25));           // pad_amount: 4 DWORDs
  g1[1] = (int)((td0 & 0xFFFFu) << 16);
  g1[2] = (int)((td0 >> 16) | ((td1 & 0xFFFFu) << 16));
  g1[3] = (int)((td1 >> 16) | ((unsigned)tileK << 16));     // tile_dim0
  g1[4] = (int)(unsigned)tileM;                             // tile_dim1
  g1[5] = (int)(unsigned)(s0 & 0xFFFFFFFFu);
  g1[6] = (int)(unsigned)((s0 >> 32) & 0xFFFFu);
  g1[7] = 0;
  i32x4 z4 = {0, 0, 0, 0};
  i32x8 z8 = {0, 0, 0, 0, 0, 0, 0, 0};
  __builtin_amdgcn_tensor_load_to_lds(g0, g1, z4, z4, z8, 0);
#else
  (void)ldsOff; (void)g; (void)K; (void)tileK; (void)tileM;
#endif
}
DEVI void tdm_wait() {
#if HAVE_TDM
  __builtin_amdgcn_s_wait_tensorcnt(0);
#endif
}

// ===========================================================================
// f32 -> bf16 elementwise convert
// ===========================================================================
__global__ __launch_bounds__(256) void cvt_kernel(const float* __restrict__ in,
                                                  bf16_t* __restrict__ out,
                                                  size_t n) {
  for (size_t i = (size_t)blockIdx.x * 256 + threadIdx.x; i < n;
       i += (size_t)gridDim.x * 256)
    out[i] = (bf16_t)in[i];
}

// ===========================================================================
// p = silu(c) @ w_ada + b_ada      (B x COND) @ (COND x 6D), tiny GEMM
// ===========================================================================
__global__ __launch_bounds__(256) void ada_kernel(const float* __restrict__ c,
                                                  const float* __restrict__ w,
                                                  const float* __restrict__ bias,
                                                  float* __restrict__ p,
                                                  int K, int Nc) {
  int j = blockIdx.x * 256 + threadIdx.x;
  int b = blockIdx.y;
  float s = 0.f;
  for (int k = 0; k < K; ++k) {
    float cv = c[(size_t)b * K + k];
    float sl = cv / (1.f + __expf(-cv));
    s += sl * w[(size_t)k * Nc + j];
  }
  p[(size_t)b * Nc + j] = s + bias[j];
}

// ===========================================================================
// Fused LayerNorm + adaLN modulation -> xn1, xn2 (bf16). One block per row.
// ===========================================================================
__global__ __launch_bounds__(256) void ln_mod_kernel(const float* __restrict__ x,
                                                     const float* __restrict__ p,
                                                     bf16_t* __restrict__ xn1,
                                                     bf16_t* __restrict__ xn2,
                                                     int D, int tokensPerB, int pLd) {
  int row = blockIdx.x;
  int b = row / tokensPerB;
  const float* xr = x + (size_t)row * D;
  float s = 0.f, s2 = 0.f;
  for (int cix = threadIdx.x; cix < D; cix += 256) {
    float v = xr[cix];
    s += v;
    s2 += v * v;
  }
  __shared__ float red[17];
  s = wave_sum(s);
  s2 = wave_sum(s2);
  int wave = threadIdx.x >> 5, lane = threadIdx.x & 31;
  if (lane == 0) { red[wave] = s; red[8 + wave] = s2; }
  __syncthreads();
  if (threadIdx.x == 0) {
    float a = 0.f, bb = 0.f;
    for (int i = 0; i < 8; ++i) { a += red[i]; bb += red[8 + i]; }
    float mean = a / D;
    red[0] = mean;
    red[1] = rsqrtf(bb / D - mean * mean + 1e-5f);
  }
  __syncthreads();
  float mean = red[0], rstd = red[1];
  const float* pb = p + (size_t)b * pLd;
  for (int cix = threadIdx.x; cix < D; cix += 256) {
    float xn = (xr[cix] - mean) * rstd;
    float v1 = xn * (1.f + pb[D + cix]) + pb[cix];             // scale1, shift1
    float v2 = xn * (1.f + pb[4 * D + cix]) + pb[3 * D + cix]; // scale2, shift2
    xn1[(size_t)row * D + cix] = (bf16_t)v1;
    xn2[(size_t)row * D + cix] = (bf16_t)v2;
  }
}

// ===========================================================================
// LayerNorm + affine (gamma, beta) -> bf16 ; one block per row
// ===========================================================================
__global__ __launch_bounds__(256) void ln_affine_kernel(const float* __restrict__ x,
                                                        const float* __restrict__ g,
                                                        const float* __restrict__ be,
                                                        bf16_t* __restrict__ out,
                                                        int D) {
  int row = blockIdx.x;
  const float* xr = x + (size_t)row * D;
  float s = 0.f, s2 = 0.f;
  for (int cix = threadIdx.x; cix < D; cix += 256) {
    float v = xr[cix];
    s += v;
    s2 += v * v;
  }
  __shared__ float red[17];
  s = wave_sum(s);
  s2 = wave_sum(s2);
  int wave = threadIdx.x >> 5, lane = threadIdx.x & 31;
  if (lane == 0) { red[wave] = s; red[8 + wave] = s2; }
  __syncthreads();
  if (threadIdx.x == 0) {
    float a = 0.f, bb = 0.f;
    for (int i = 0; i < 8; ++i) { a += red[i]; bb += red[8 + i]; }
    float mean = a / D;
    red[0] = mean;
    red[1] = rsqrtf(bb / D - mean * mean + 1e-5f);
  }
  __syncthreads();
  float mean = red[0], rstd = red[1];
  for (int cix = threadIdx.x; cix < D; cix += 256) {
    float xn = (xr[cix] - mean) * rstd;
    out[(size_t)row * D + cix] = (bf16_t)(xn * g[cix] + be[cix]);
  }
}

// ===========================================================================
// Generic bf16 WMMA GEMM: C(MxN) = A(MxK) * B(KxN)  (+ epilogue)
// BM=128 BN=128 BK=32 ; 256 threads = 8 waves (4x2), 32x64 per wave
// => 8 v_wmma per wave per k-step.
// GUARD=false (M % 128 == 0): branch-free inner loop, A tiles staged by the
//   Tensor Data Mover (async, ping-pong LDS buffers, s_wait_tensorcnt).
// GUARD=true: manual guarded A staging (used for the M=308 KV projection).
// epi: 0 -> outH = bf16(acc)
//      1 -> outH = bf16(gelu(acc + bias))
//      2 -> outF = resid + gate * (acc [+ bias])   (gate null -> 1.0)
// ===========================================================================
#define GEMM_BM 128
#define GEMM_BN 128
#define GEMM_BK 32
#define GEMM_PA 40   // bf16 pitch; 80B row stride (16B multiple, TDM pad)
#define GEMM_PB 34   // bf16 pitch (even)

DEVI void stage_b_tile(const bf16_t* __restrict__ Bw, bf16_t* Bs, int N,
                       int col0, int kt, int tid) {
#pragma unroll
  for (int i = 0; i < 2; ++i) {
    int c = tid + i * 256;                  // 32 k-rows x 16 chunks
    int k = c >> 4, nc = (c & 15) * 8;
    uint4 v = *(const uint4*)(Bw + (size_t)(kt + k) * N + col0 + nc);
    const bf16_t* pv = (const bf16_t*)&v;
#pragma unroll
    for (int j = 0; j < 8; ++j) Bs[(nc + j) * GEMM_PB + k] = pv[j];
  }
}

DEVI void stage_a_tile_guard(const bf16_t* __restrict__ A, bf16_t* As, int M,
                             int K, int row0, int kt, int tid) {
#pragma unroll
  for (int i = 0; i < 2; ++i) {
    int c = tid + i * 256;                  // 128 rows x 4 chunks
    int r = c >> 2, kc = (c & 3) * 8;
    uint4 v = make_uint4(0, 0, 0, 0);
    if (row0 + r < M) v = *(const uint4*)(A + (size_t)(row0 + r) * K + kt + kc);
    *(uint4*)&As[r * GEMM_PA + kc] = v;
  }
}

template <bool GUARD>
__global__ __launch_bounds__(256) void gemm_bf16_wmma(
    const bf16_t* __restrict__ A, const bf16_t* __restrict__ Bw,
    int M, int N, int K,
    float* __restrict__ outF, bf16_t* __restrict__ outH,
    const float* __restrict__ resid, const float* __restrict__ bias,
    const float* __restrict__ gate, int gateLd, int rowsPerB, int epi) {
  __shared__ bf16_t As[2][GEMM_BM * GEMM_PA];
  __shared__ bf16_t Bs[GEMM_BN * GEMM_PB];

  int tid = threadIdx.x;
  int lane = tid & 31, wave = tid >> 5;
  int row0 = blockIdx.x * GEMM_BM;
  int col0 = blockIdx.y * GEMM_BN;
  int wm = wave >> 1, wn = wave & 1;  // 4x2 wave grid: 32 rows x 64 cols

  v8f acc[2][4];
#pragma unroll
  for (int i = 0; i < 2; ++i)
#pragma unroll
    for (int j = 0; j < 4; ++j) acc[i][j] = (v8f){0, 0, 0, 0, 0, 0, 0, 0};

  constexpr bool kTdmAvail = (HAVE_TDM != 0);
  const bool useTdm = kTdmAvail && !GUARD;
  const int nk = K / GEMM_BK;

  // ---- prologue: stage B(0); A(0) via TDM or manual ----
  stage_b_tile(Bw, Bs, N, col0, 0, tid);
  if (useTdm) {
    if (wave == 0) {
      tdm_load_a_tile((unsigned)(uintptr_t)(void*)&As[0][0],
                      A + (size_t)row0 * K, K, GEMM_BK, GEMM_BM);
      tdm_wait();
    }
  } else {
    stage_a_tile_guard(A, &As[0][0], M, K, row0, 0, tid);
  }
  __syncthreads();

  for (int it = 0; it < nk; ++it) {
    int cur = it & 1;
    // async: kick DMA for next A tile into the other buffer
    if (useTdm && wave == 0 && it + 1 < nk)
      tdm_load_a_tile((unsigned)(uintptr_t)(void*)&As[cur ^ 1][0],
                      A + (size_t)row0 * K + (size_t)(it + 1) * GEMM_BK, K,
                      GEMM_BK, GEMM_BM);
    // prefetch next B tile into L2
    if (it + 1 < nk)
      __builtin_prefetch(Bw + (size_t)((it + 1) * GEMM_BK + (tid >> 4)) * N +
                             col0 + (tid & 15) * 8, 0, 0);

    const bf16_t* Ac = &As[cur][0];
    v16bf a0 = load_frag_a(Ac, GEMM_PA, wm * 32, 0, lane);
    v16bf a1 = load_frag_a(Ac, GEMM_PA, wm * 32 + 16, 0, lane);
#pragma unroll
    for (int sn = 0; sn < 4; ++sn) {
      v16bf b = load_frag_b(Bs, GEMM_PB, wn * 64 + sn * 16, 0, lane);
      acc[0][sn] = wmma_bf16(a0, b, acc[0][sn]);
      acc[1][sn] = wmma_bf16(a1, b, acc[1][sn]);
    }
    __syncthreads();  // everyone done reading Bs (and As[cur])

    if (it + 1 < nk) {
      stage_b_tile(Bw, Bs, N, col0, (it + 1) * GEMM_BK, tid);
      if (useTdm) {
        if (wave == 0) tdm_wait();
      } else {
        stage_a_tile_guard(A, &As[(it + 1) & 1][0], M, K, row0,
                           (it + 1) * GEMM_BK, tid);
      }
      __syncthreads();
    }
  }

  // ---- epilogue (C layout: VGPR r -> M = r + hi*8, N = lane&15) ----
  int lo = lane & 15, hi = lane >> 4;
#pragma unroll
  for (int sm = 0; sm < 2; ++sm)
#pragma unroll
    for (int sn = 0; sn < 4; ++sn)
#pragma unroll
      for (int r = 0; r < 8; ++r) {
        int row = row0 + wm * 32 + sm * 16 + hi * 8 + r;
        int col = col0 + wn * 64 + sn * 16 + lo;
        if (GUARD && row >= M) continue;
        float a = acc[sm][sn][r];
        size_t idx = (size_t)row * N + col;
        if (epi == 0) {
          outH[idx] = (bf16_t)a;
        } else if (epi == 1) {
          float t = a + bias[col];
          float gl = 0.5f * t * (1.f + erff(t * 0.70710678118654752f));
          outH[idx] = (bf16_t)gl;
        } else {  // epi == 2: residual (+bias) (*gate)
          float t = bias ? (a + bias[col]) : a;
          float gv = gate ? gate[(size_t)(row / rowsPerB) * gateLd + col] : 1.0f;
          outF[idx] = resid[idx] + gv * t;
        }
      }
}

// ===========================================================================
// Fused self-attention, one (b,h) x 64-query panel per block.
// Full 64x1024 f32 score panel resident in WGP LDS (dynamic, ~283KB).
// qkv layout: (B,N,3*H*Dh) with q|k|v at col offsets 0|D|2D.
// ===========================================================================
#define SA_SP 1032  // f32 score pitch (even)
#define SA_QP 72    // bf16 pitch (144B, 16B multiple)

__global__ __launch_bounds__(256) void self_attn_kernel(
    const bf16_t* __restrict__ qkv, bf16_t* __restrict__ sa,
    int Nn, int Hh, float scale) {
  extern __shared__ char smem_raw[];
  float* S = (float*)smem_raw;                               // 64 x 1032 f32
  bf16_t* Qs = (bf16_t*)(smem_raw + 64 * SA_SP * 4);         // 64 x 72 bf16
  bf16_t* KVs = Qs + 64 * SA_QP;                             // 64 x 72 bf16

  int bh = blockIdx.x;
  int b = bh / Hh, h = bh % Hh;
  int n0 = blockIdx.y * 64;
  int tid = threadIdx.x, lane = tid & 31, wave = tid >> 5;
  int qvRow = tid >> 2, qvChunk = (tid & 3) * 16;
  size_t qkvPitch = (size_t)3 * Hh * 64;

  {  // Q panel (64 x 64 bf16)
    const uint4* src = (const uint4*)(qkv + ((size_t)(b * Nn + n0 + qvRow)) * qkvPitch +
                                      h * 64 + qvChunk);
    uint4 v0 = src[0], v1 = src[1];
    *(uint4*)&Qs[qvRow * SA_QP + qvChunk] = v0;
    *(uint4*)&Qs[qvRow * SA_QP + qvChunk + 8] = v1;
  }

  // ---- phase 1: S = scale * Q K^T over all key blocks ----
  for (int j = 0; j < Nn / 64; ++j) {
    {  // K tile, n-major naturally: KVs[key][dh]
      const uint4* src = (const uint4*)(qkv + ((size_t)(b * Nn + j * 64 + qvRow)) * qkvPitch +
                                        Hh * 64 + h * 64 + qvChunk);
      uint4 v0 = src[0], v1 = src[1];
      *(uint4*)&KVs[qvRow * SA_QP + qvChunk] = v0;
      *(uint4*)&KVs[qvRow * SA_QP + qvChunk + 8] = v1;
    }
    __syncthreads();
#pragma unroll
    for (int i = 0; i < 2; ++i) {
      int t2 = wave * 2 + i;        // 16 tiles: 4 qsub x 4 ksub
      int qs = t2 >> 2, ks = t2 & 3;
      v8f a = (v8f){0, 0, 0, 0, 0, 0, 0, 0};
      v16bf fa0 = load_frag_a(Qs, SA_QP, qs * 16, 0, lane);
      v16bf fa1 = load_frag_a(Qs, SA_QP, qs * 16, 32, lane);
      v16bf fb0 = load_frag_b(KVs, SA_QP, ks * 16, 0, lane);
      v16bf fb1 = load_frag_b(KVs, SA_QP, ks * 16, 32, lane);
      a = wmma_bf16(fa0, fb0, a);
      a = wmma_bf16(fa1, fb1, a);
      int lo = lane & 15, hi = lane >> 4;
#pragma unroll
      for (int r = 0; r < 8; ++r)
        S[(qs * 16 + hi * 8 + r) * SA_SP + j * 64 + ks * 16 + lo] = a[r] * scale;
    }
    __syncthreads();
  }

  // ---- phase 2: row softmax over 1024 keys (8 rows per wave) ----
  for (int rr = 0; rr < 8; ++rr) {
    int row = wave * 8 + rr;
    float* Sr = S + (size_t)row * SA_SP;
    float mx = -INFINITY;
    for (int cix = lane; cix < Nn; cix += 32) mx = fmaxf(mx, Sr[cix]);
    mx = wave_max(mx);
    float sm = 0.f;
    for (int cix = lane; cix < Nn; cix += 32) {
      float e = __expf(Sr[cix] - mx);
      Sr[cix] = e;
      sm += e;
    }
    sm = wave_sum(sm);
    float inv = 1.f / sm;
    for (int cix = lane; cix < Nn; cix += 32) Sr[cix] *= inv;
  }
  __syncthreads();

  // ---- phase 3: O = P V ----
  v8f o[2];
  o[0] = (v8f){0, 0, 0, 0, 0, 0, 0, 0};
  o[1] = (v8f){0, 0, 0, 0, 0, 0, 0, 0};
  for (int j = 0; j < Nn / 64; ++j) {
    {  // V tile transposed -> KVs[dh][key]
      __align__(16) bf16_t tmp[16];
      const uint4* src = (const uint4*)(qkv + ((size_t)(b * Nn + j * 64 + qvRow)) * qkvPitch +
                                        2 * Hh * 64 + h * 64 + qvChunk);
      *(uint4*)&tmp[0] = src[0];
      *(uint4*)&tmp[8] = src[1];
#pragma unroll
      for (int e = 0; e < 16; ++e) KVs[(qvChunk + e) * SA_QP + qvRow] = tmp[e];
    }
    __syncthreads();
#pragma unroll
    for (int i = 0; i < 2; ++i) {
      int t2 = wave * 2 + i;        // 16 tiles: 4 qsub x 4 dsub
      int qs = t2 >> 2, ds = t2 & 3;
      v16bf fa0 = load_frag_a_f32(S, SA_SP, qs * 16, j * 64, lane);
      v16bf fa1 = load_frag_a_f32(S, SA_SP, qs * 16, j * 64 + 32, lane);
      v16bf fb0 = load_frag_b(KVs, SA_QP, ds * 16, 0, lane);
      v16bf fb1 = load_frag_b(KVs, SA_QP, ds * 16, 32, lane);
      o[i] = wmma_bf16(fa0, fb0, o[i]);
      o[i] = wmma_bf16(fa1, fb1, o[i]);
    }
    __syncthreads();
  }
  // ---- store O -> sa (B,N,D) bf16 ----
  int lo = lane & 15, hi = lane >> 4;
#pragma unroll
  for (int i = 0; i < 2; ++i) {
    int t2 = wave * 2 + i;
    int qs = t2 >> 2, ds = t2 & 3;
#pragma unroll
    for (int r = 0; r < 8; ++r) {
      int row = n0 + qs * 16 + hi * 8 + r;
      int col = h * 64 + ds * 16 + lo;
      sa[((size_t)(b * Nn + row)) * (Hh * 64) + col] = (bf16_t)o[i][r];
    }
  }
}

// ===========================================================================
// Fused cross-attention (T=77 keys padded to 80/96). Writes attn_w (f32, to
// d_out) and ca (bf16). One (b,h) x 64-query panel per block. ~60KB static LDS.
// qc: (B,N,H*Dh) bf16 ; kvc: (B,T,2*H*Dh) bf16 (k then v halves)
// ===========================================================================
#define CA_SP 104  // pitches (even)
#define CA_QP 72

__global__ __launch_bounds__(256) void cross_attn_kernel(
    const bf16_t* __restrict__ qc, const bf16_t* __restrict__ kvc,
    const unsigned char* __restrict__ mask,
    float* __restrict__ attnw, bf16_t* __restrict__ ca,
    int Nn, int Tt, int Hh, float scale) {
  __shared__ bf16_t Qs[64 * CA_QP];
  __shared__ bf16_t Ks[80 * CA_QP];   // n-major: Ks[key][dh]
  __shared__ bf16_t Vs[64 * CA_SP];   // n-major: Vs[dh][key], keys padded to 96
  __shared__ float S[64 * CA_SP];     // scores / probs

  int bh = blockIdx.x;
  int b = bh / Hh, h = bh % Hh;
  int n0 = blockIdx.y * 64;
  int tid = threadIdx.x, lane = tid & 31, wave = tid >> 5;
  size_t kvPitch = (size_t)2 * Hh * 64;

  {  // Q panel
    int row = tid >> 2, ch = (tid & 3) * 16;
    const uint4* src = (const uint4*)(qc + ((size_t)(b * Nn + n0 + row)) * (Hh * 64) +
                                      h * 64 + ch);
    uint4 v0 = src[0], v1 = src[1];
    *(uint4*)&Qs[row * CA_QP + ch] = v0;
    *(uint4*)&Qs[row * CA_QP + ch + 8] = v1;
  }
  // K rows 0..79 (zero pad beyond T)
  for (int c = tid; c < 80 * 4; c += 256) {
    int t = c >> 2, ch = (c & 3) * 16;
    uint4 v0 = make_uint4(0, 0, 0, 0), v1 = make_uint4(0, 0, 0, 0);
    if (t < Tt) {
      const uint4* src = (const uint4*)(kvc + ((size_t)(b * Tt + t)) * kvPitch + h * 64 + ch);
      v0 = src[0];
      v1 = src[1];
    }
    *(uint4*)&Ks[t * CA_QP + ch] = v0;
    *(uint4*)&Ks[t * CA_QP + ch + 8] = v1;
  }
  // V transposed, keys 0..95 (zero pad)
  for (int c = tid; c < 96 * 4; c += 256) {
    int t = c >> 2, ch = (c & 3) * 16;
    __align__(16) bf16_t tmp[16];
    if (t < Tt) {
      const uint4* src = (const uint4*)(kvc + ((size_t)(b * Tt + t)) * kvPitch +
                                        Hh * 64 + h * 64 + ch);
      *(uint4*)&tmp[0] = src[0];
      *(uint4*)&tmp[8] = src[1];
    } else {
      *(uint4*)&tmp[0] = make_uint4(0, 0, 0, 0);
      *(uint4*)&tmp[8] = make_uint4(0, 0, 0, 0);
    }
#pragma unroll
    for (int e = 0; e < 16; ++e) Vs[(ch + e) * CA_SP + t] = tmp[e];
  }
  __syncthreads();

  // ---- logits: 20 tiles (4 qsub x 5 ksub) ----
  for (int t2 = wave; t2 < 20; t2 += 8) {
    int qs = t2 & 3, ks = t2 >> 2;
    v8f a = (v8f){0, 0, 0, 0, 0, 0, 0, 0};
    v16bf fa0 = load_frag_a(Qs, CA_QP, qs * 16, 0, lane);
    v16bf fa1 = load_frag_a(Qs, CA_QP, qs * 16, 32, lane);
    v16bf fb0 = load_frag_b(Ks, CA_QP, ks * 16, 0, lane);
    v16bf fb1 = load_frag_b(Ks, CA_QP, ks * 16, 32, lane);
    a = wmma_bf16(fa0, fb0, a);
    a = wmma_bf16(fa1, fb1, a);
    int lo = lane & 15, hi = lane >> 4;
#pragma unroll
    for (int r = 0; r < 8; ++r)
      S[(qs * 16 + hi * 8 + r) * CA_SP + ks * 16 + lo] = a[r] * scale;
  }
  __syncthreads();

  // ---- masked softmax over T, write attn_w ----
  for (int rr = 0; rr < 8; ++rr) {
    int row = wave * 8 + rr;
    float vals[3];
    float mx = -INFINITY;
#pragma unroll
    for (int i = 0; i < 3; ++i) {
      int cix = lane + i * 32;
      float v = -INFINITY;
      if (cix < Tt && !mask[(size_t)b * Tt + cix]) v = S[row * CA_SP + cix];
      vals[i] = v;
      mx = fmaxf(mx, v);
    }
    mx = wave_max(mx);
    float sm = 0.f;
#pragma unroll
    for (int i = 0; i < 3; ++i) {
      float e = (vals[i] == -INFINITY) ? 0.f : __expf(vals[i] - mx);
      vals[i] = e;
      sm += e;
    }
    sm = wave_sum(sm);
    float inv = 1.f / sm;
#pragma unroll
    for (int i = 0; i < 3; ++i) {
      int cix = lane + i * 32;
      float pv = vals[i] * inv;
      S[row * CA_SP + cix] = pv;  // padded cols become 0
      if (cix < Tt)
        attnw[(((size_t)(b * Hh + h) * Nn) + n0 + row) * Tt + cix] = pv;
    }
  }
  __syncthreads();

  // ---- ca = P V : 16 tiles (4 qsub x 4 dsub) ----
  int lo = lane & 15, hi = lane >> 4;
#pragma unroll
  for (int i = 0; i < 2; ++i) {
    int t2 = wave * 2 + i;
    int qs = t2 >> 2, ds = t2 & 3;
    v8f o = (v8f){0, 0, 0, 0, 0, 0, 0, 0};
#pragma unroll
    for (int k0 = 0; k0 < 96; k0 += 32) {
      v16bf fa = load_frag_a_f32(S, CA_SP, qs * 16, k0, lane);
      v16bf fb = load_frag_b(Vs, CA_SP, ds * 16, k0, lane);
      o = wmma_bf16(fa, fb, o);
    }
#pragma unroll
    for (int r = 0; r < 8; ++r) {
      int row = n0 + qs * 16 + hi * 8 + r;
      int col = h * 64 + ds * 16 + lo;
      ca[((size_t)(b * Nn + row)) * (Hh * 64) + col] = (bf16_t)o[r];
    }
  }
}

// ===========================================================================
// Host orchestration
// ===========================================================================
extern "C" void kernel_launch(void* const* d_in, const int* in_sizes, int n_in,
                              void* d_out, int out_size, void* d_ws, size_t ws_size,
                              hipStream_t stream) {
  (void)in_sizes; (void)n_in; (void)out_size; (void)ws_size;
  const int B = 4, N = 1024, T = 77, D = 1024, H = 16, DFF = 4096, COND = 1024;
  const float scale = 0.125f;  // Dh^-0.5, Dh=64

  const float* x      = (const float*)d_in[0];
  const float* c      = (const float*)d_in[1];
  const float* ctx    = (const float*)d_in[2];
  const unsigned char* ctx_mask = (const unsigned char*)d_in[3];
  const float* w_ada  = (const float*)d_in[4];
  const float* b_ada  = (const float*)d_in[5];
  const float* w_qkv  = (const float*)d_in[6];
  const float* w_sa   = (const float*)d_in[7];
  const float* w_q    = (const float*)d_in[8];
  const float* w_kv   = (const float*)d_in[9];
  const float* w_ca   = (const float*)d_in[10];
  const float* g_ca   = (const float*)d_in[11];
  const float* b_ca   = (const float*)d_in[12];
  const float* g_ctx  = (const float*)d_in[13];
  const float* b_ctx  = (const float*)d_in[14];
  const float* w1     = (const float*)d_in[15];
  const float* b1     = (const float*)d_in[16];
  const float* w2     = (const float*)d_in[17];
  const float* b2     = (const float*)d_in[18];

  float* xOut = (float*)d_out;                              // (B,N,D)
  float* attnwOut = (float*)d_out + (size_t)B * N * D;      // (B,H,N,T)

  // ---- workspace carve-up ----
  size_t off = 0;
  auto take = [&](size_t bytes) {
    size_t o = off;
    off = (off + bytes + 255) & ~(size_t)255;
    return o;
  };
  char* ws = (char*)d_ws;
  float*  p_buf  = (float*)(ws + take((size_t)B * 6 * D * 4));
  bf16_t* xn1    = (bf16_t*)(ws + take((size_t)B * N * D * 2));
  bf16_t* xn2    = (bf16_t*)(ws + take((size_t)B * N * D * 2));
  bf16_t* wqkv_h = (bf16_t*)(ws + take((size_t)D * 3 * D * 2));
  bf16_t* wsa_h  = (bf16_t*)(ws + take((size_t)D * D * 2));
  bf16_t* wq_h   = (bf16_t*)(ws + take((size_t)D * D * 2));
  bf16_t* wkv_h  = (bf16_t*)(ws + take((size_t)D * 2 * D * 2));
  bf16_t* wca_h  = (bf16_t*)(ws + take((size_t)D * D * 2));
  bf16_t* w1_h   = (bf16_t*)(ws + take((size_t)D * DFF * 2));
  bf16_t* w2_h   = (bf16_t*)(ws + take((size_t)DFF * D * 2));
  bf16_t* qkv    = (bf16_t*)(ws + take((size_t)B * N * 3 * D * 2));
  bf16_t* sa     = (bf16_t*)(ws + take((size_t)B * N * D * 2));
  bf16_t* qy     = (bf16_t*)(ws + take((size_t)B * N * D * 2));
  bf16_t* cx     = (bf16_t*)(ws + take((size_t)B * T * D * 2));
  bf16_t* qc     = (bf16_t*)(ws + take((size_t)B * N * D * 2));
  bf16_t* kvc    = (bf16_t*)(ws + take((size_t)B * T * 2 * D * 2));
  bf16_t* ca     = (bf16_t*)(ws + take((size_t)B * N * D * 2));
  bf16_t* hbuf   = (bf16_t*)(ws + take((size_t)B * N * DFF * 2));

  // ---- weight converts f32 -> bf16 ----
  auto cvt = [&](const float* src, bf16_t* dst, size_t n) {
    cvt_kernel<<<2048, 256, 0, stream>>>(src, dst, n);
  };
  cvt(w_qkv, wqkv_h, (size_t)D * 3 * D);
  cvt(w_sa, wsa_h, (size_t)D * D);
  cvt(w_q, wq_h, (size_t)D * D);
  cvt(w_kv, wkv_h, (size_t)D * 2 * D);
  cvt(w_ca, wca_h, (size_t)D * D);
  cvt(w1, w1_h, (size_t)D * DFF);
  cvt(w2, w2_h, (size_t)DFF * D);

  // ---- adaLN modulation: p = silu(c) @ w_ada + b_ada ----
  ada_kernel<<<dim3(6 * D / 256, B), 256, 0, stream>>>(c, w_ada, b_ada, p_buf, COND, 6 * D);

  // ---- fused LN + modulation -> xn1, xn2 (bf16) ----
  ln_mod_kernel<<<B * N, 256, 0, stream>>>(x, p_buf, xn1, xn2, D, N, 6 * D);

  // ---- QKV projection: (4096x1024)@(1024x3072) ----
  gemm_bf16_wmma<false><<<dim3(B * N / GEMM_BM, 3 * D / GEMM_BN), 256, 0, stream>>>(
      xn1, wqkv_h, B * N, 3 * D, D, nullptr, qkv, nullptr, nullptr, nullptr, 0, N, 0);

  // ---- self-attention (full score panel in WGP LDS) ----
  {
    size_t smBytes = (size_t)64 * SA_SP * 4 + (size_t)2 * 64 * SA_QP * 2;
    self_attn_kernel<<<dim3(B * H, N / 64), 256, smBytes, stream>>>(qkv, sa, N, H, scale);
  }

  // ---- SA projection + gated residual: x1 = x + gate1 * (sa @ w_sa_proj) ----
  gemm_bf16_wmma<false><<<dim3(B * N / GEMM_BM, D / GEMM_BN), 256, 0, stream>>>(
      sa, wsa_h, B * N, D, D, xOut, nullptr, x, nullptr, p_buf + 2 * D, 6 * D, N, 2);

  // ---- cross-attn pre-norms ----
  ln_affine_kernel<<<B * N, 256, 0, stream>>>(xOut, g_ca, b_ca, qy, D);
  ln_affine_kernel<<<B * T, 256, 0, stream>>>(ctx, g_ctx, b_ctx, cx, D);

  // ---- q / kv projections ----
  gemm_bf16_wmma<false><<<dim3(B * N / GEMM_BM, D / GEMM_BN), 256, 0, stream>>>(
      qy, wq_h, B * N, D, D, nullptr, qc, nullptr, nullptr, nullptr, 0, N, 0);
  gemm_bf16_wmma<true><<<dim3((B * T + GEMM_BM - 1) / GEMM_BM, 2 * D / GEMM_BN), 256, 0, stream>>>(
      cx, wkv_h, B * T, 2 * D, D, nullptr, kvc, nullptr, nullptr, nullptr, 0, T, 0);

  // ---- fused cross-attention (writes attn_w to d_out) ----
  cross_attn_kernel<<<dim3(B * H, N / 64), 256, 0, stream>>>(
      qc, kvc, ctx_mask, attnwOut, ca, N, T, H, scale);

  // ---- CA out projection + residual: x2 = x1 + ca @ w_ca_out ----
  gemm_bf16_wmma<false><<<dim3(B * N / GEMM_BM, D / GEMM_BN), 256, 0, stream>>>(
      ca, wca_h, B * N, D, D, xOut, nullptr, xOut, nullptr, nullptr, 0, N, 2);

  // ---- FFN: h = gelu(xn2 @ w1 + b1) ; x3 = x2 + gate2 * (h @ w2 + b2) ----
  gemm_bf16_wmma<false><<<dim3(B * N / GEMM_BM, DFF / GEMM_BN), 256, 0, stream>>>(
      xn2, w1_h, B * N, DFF, D, nullptr, hbuf, nullptr, b1, nullptr, 0, N, 1);
  gemm_bf16_wmma<false><<<dim3(B * N / GEMM_BM, D / GEMM_BN), 256, 0, stream>>>(
      hbuf, w2_h, B * N, D, DFF, xOut, nullptr, xOut, b2, p_buf + 5 * D, 6 * D, N, 2);
}
